// SpatioTemporalGNN_45226005627008
// MI455X (gfx1250) — compile-verified, matrix operand order checked by reading, not run on hardware
//
#include <hip/hip_runtime.h>
#include <hip/hip_bf16.h>
#include <math.h>

typedef __attribute__((ext_vector_type(16))) _Float16 v16h;
typedef __attribute__((ext_vector_type(8)))  _Float16 v8h;
typedef __attribute__((ext_vector_type(8)))  float    v8f;

#define GH 32
#define RH 64
#define STAT 6
#define DYNF 1

// ---------------------------------------------------------------------------
// WMMA helper (f16 in, f32 accumulate), wave32, 16x16 tiles, K=32 per op.
// ---------------------------------------------------------------------------
__device__ __forceinline__ v8f wmma_f16(v16h a, v16h b, v8f c) {
  return __builtin_amdgcn_wmma_f32_16x16x32_f16(
      /*neg_a=*/false, a, /*neg_b=*/false, b,
      /*c_mod=*/(short)0, c, /*reuse_a=*/false, /*reuse_b=*/false);
}

// Per the ISA 16-bit 16x32 fragment layout, each lane's 16 halfs are two
// contiguous 8-half (16B) runs: K = kh+0..7 (VGPR0-3) and K = kh+16..23
// (VGPR4-7), kh = k0 + 8*(lane/16). Fetch each run as one b128 load.
__device__ __forceinline__ v16h combine16(v8h lo, v8h hi) {
  return __builtin_shufflevector(lo, hi, 0, 1, 2, 3, 4, 5, 6, 7,
                                 8, 9, 10, 11, 12, 13, 14, 15);
}

// A fragment: 16 rows (row0..+15) x 32 K (k0..+31) from row-major M[ld].
__device__ __forceinline__ v16h load_a_frag(const _Float16* M, int row0, int ld,
                                            int k0, int lane) {
  int r  = row0 + (lane & 15);
  int kh = k0 + ((lane >> 4) << 3);
  const _Float16* p = M + (size_t)r * ld + kh;
  v8h lo = *(const v8h*)(p);
  v8h hi = *(const v8h*)(p + 16);
  return combine16(lo, hi);
}

// B fragment where B[k, j] = W[j*ldw + k]  (GEMM with W^T, torch-style
// [out,in] row-major weight). Same two-b128 pattern along the K axis.
__device__ __forceinline__ v16h load_bT_frag(const _Float16* W, int col0, int ldw,
                                             int k0, int lane) {
  int c  = col0 + (lane & 15);
  int kh = k0 + ((lane >> 4) << 3);
  const _Float16* p = W + (size_t)c * ldw + kh;
  v8h lo = *(const v8h*)(p);
  v8h hi = *(const v8h*)(p + 16);
  return combine16(lo, hi);
}

// ---------------------------------------------------------------------------
// Utility kernels
// ---------------------------------------------------------------------------
__global__ void k_fill_f32(float* p, float v, int n) {
  int i = blockIdx.x * blockDim.x + threadIdx.x;
  if (i < n) p[i] = v;
}

__global__ void k_fill_f16_zero(_Float16* p, int n) {
  int i = blockIdx.x * blockDim.x + threadIdx.x;
  if (i < n) p[i] = (_Float16)0.0f;
}

__global__ void k_cvt_f16(const float* s, _Float16* d, int n) {
  int i = blockIdx.x * blockDim.x + threadIdx.x;
  if (i < n) d[i] = (_Float16)s[i];
}

// Convert + transpose: d[j*K + k] = (f16) s[k*N + j]   (s is [K,N] row-major)
__global__ void k_cvt_f16_T(const float* s, _Float16* d, int K, int Nc) {
  int i = blockIdx.x * blockDim.x + threadIdx.x;
  if (i >= K * Nc) return;
  int j = i / K, k = i % K;
  d[j * K + k] = (_Float16)s[k * Nc + j];
}

// deg: prefilled with 1.0 (self loop). Add 1 per incoming edge.
__global__ void k_degree(const int* ei, float* deg, int E) {
  int e = blockIdx.x * blockDim.x + threadIdx.x;
  if (e < E) atomicAdd(&deg[ei[E + e]], 1.0f);
}

__global__ void k_dinv(float* deg, int N) {
  int i = blockIdx.x * blockDim.x + threadIdx.x;
  if (i < N) deg[i] = rsqrtf(deg[i]);  // deg >= 1 always
}

// ---------------------------------------------------------------------------
// GCN layer 0 transform: pre0[i,:] = [dyn[t,i], static[i,:]] @ w0  (7 -> 32)
// ---------------------------------------------------------------------------
__global__ void k_gcn_transform0(const float* __restrict__ dyn,
                                 const float* __restrict__ stat,
                                 const float* __restrict__ w0,
                                 float* __restrict__ pre, int N, int t) {
  __shared__ float w0s[(DYNF + STAT) * GH];
  if (threadIdx.x < (DYNF + STAT) * GH) w0s[threadIdx.x] = w0[threadIdx.x];
  __syncthreads();
  int i = blockIdx.x * blockDim.x + threadIdx.x;
  if (i >= N) return;
  float x[DYNF + STAT];
  x[0] = dyn[(size_t)t * N + i];
#pragma unroll
  for (int k = 0; k < STAT; ++k) x[1 + k] = stat[(size_t)i * STAT + k];
  float* po = pre + (size_t)i * GH;
#pragma unroll 4
  for (int f = 0; f < GH; ++f) {
    float acc = 0.0f;
#pragma unroll
    for (int k = 0; k < DYNF + STAT; ++k) acc = fmaf(x[k], w0s[k * GH + f], acc);
    po[f] = acc;
  }
}

// ---------------------------------------------------------------------------
// GCN layer 1 transform (WMMA): pre1 = h0_f16 @ w1   ([N,32] @ [32,32]),
// w1 pre-transposed to [out,in] so the coalesced bT loader applies.
// ---------------------------------------------------------------------------
__global__ void k_gcn_transform1(const _Float16* __restrict__ hin,
                                 const _Float16* __restrict__ w1T,
                                 float* __restrict__ pre, int N) {
  int lane = threadIdx.x & 31;
  int tile = blockIdx.x * (blockDim.x >> 5) + (threadIdx.x >> 5);
  int m0 = tile * 16;
  if (m0 >= N) return;
  int col16 = lane & 15, half = lane >> 4;
  v16h a = load_a_frag(hin, m0, GH, 0, lane);
#pragma unroll
  for (int j = 0; j < 2; ++j) {
    v8f c = {};
    c = wmma_f16(a, load_bT_frag(w1T, j * 16, GH, 0, lane), c);
    int col = j * 16 + col16;
#pragma unroll
    for (int v = 0; v < 8; ++v) {
      int row = m0 + 8 * half + v;
      pre[(size_t)row * GH + col] = c[v];
    }
  }
}

// ---------------------------------------------------------------------------
// Edge aggregation: agg[dst,:] += dinv[src]*dinv[dst] * pre[src,:]
// One thread per (edge, 8-feature group). All traffic L2-resident.
// ---------------------------------------------------------------------------
__global__ void k_gcn_aggregate(const int* __restrict__ ei,
                                const float* __restrict__ dinv,
                                const float* __restrict__ pre,
                                float* __restrict__ agg, int E) {
  int tid = blockIdx.x * blockDim.x + threadIdx.x;
  if (tid >= E * 4) return;
  int e = tid >> 2;
  int g = (tid & 3) * 8;
  int s = ei[e];
  int d = ei[E + e];
  float nrm = dinv[s] * dinv[d];
  const float* ps = pre + (size_t)s * GH + g;
  float* pa = agg + (size_t)d * GH + g;
#pragma unroll
  for (int q = 0; q < 8; ++q) atomicAdd(pa + q, nrm * ps[q]);
}

// Finish: out = relu(agg + dinv^2*pre + bias), emitted as f16 for next WMMA.
__global__ void k_gcn_finish(const float* __restrict__ agg,
                             const float* __restrict__ pre,
                             const float* __restrict__ dinv,
                             const float* __restrict__ bias,
                             _Float16* __restrict__ out16, int N) {
  int tid = blockIdx.x * blockDim.x + threadIdx.x;
  if (tid >= N * GH) return;
  int i = tid >> 5, f = tid & 31;
  float sn = dinv[i];
  sn *= sn;  // self-loop norm
  float v = agg[tid] + sn * pre[tid] + bias[f];
  out16[tid] = (_Float16)fmaxf(v, 0.0f);
}

// ---------------------------------------------------------------------------
// GRU step (WMMA): one wave per 16-node tile. For each group of 16 hidden
// columns j: r/z pre-activations = x@Wih^T + h@Whh^T (3 WMMAs each, K=32+64),
// xn (1 WMMA), hn (2 WMMAs) kept separate for cand = tanh(xn + r*hn).
// 36 WMMAs per wave. Updates h in f32 (recurrence) and f16 (next A-frags).
// ---------------------------------------------------------------------------
__global__ void k_gru_step(const _Float16* __restrict__ xt,   // [N,32]
                           const _Float16* __restrict__ wih,  // [192,32]
                           const _Float16* __restrict__ whh,  // [192,64]
                           const float* __restrict__ bih,
                           const float* __restrict__ bhh,
                           float* __restrict__ h32,
                           _Float16* __restrict__ h16, int N) {
  int lane = threadIdx.x & 31;
  int tile = blockIdx.x * (blockDim.x >> 5) + (threadIdx.x >> 5);
  int m0 = tile * 16;
  if (m0 >= N) return;
  int col16 = lane & 15, half = lane >> 4;

  v16h ax  = load_a_frag(xt, m0, GH, 0, lane);
  v16h ah0 = load_a_frag(h16, m0, RH, 0, lane);
  v16h ah1 = load_a_frag(h16, m0, RH, 32, lane);

#pragma unroll
  for (int j = 0; j < 4; ++j) {
    int c = j * 16;
    v8f sr = {};
    sr = wmma_f16(ax,  load_bT_frag(wih, c, GH, 0, lane), sr);
    sr = wmma_f16(ah0, load_bT_frag(whh, c, RH, 0, lane), sr);
    sr = wmma_f16(ah1, load_bT_frag(whh, c, RH, 32, lane), sr);
    v8f sz = {};
    sz = wmma_f16(ax,  load_bT_frag(wih, RH + c, GH, 0, lane), sz);
    sz = wmma_f16(ah0, load_bT_frag(whh, RH + c, RH, 0, lane), sz);
    sz = wmma_f16(ah1, load_bT_frag(whh, RH + c, RH, 32, lane), sz);
    v8f xn = {};
    xn = wmma_f16(ax,  load_bT_frag(wih, 2 * RH + c, GH, 0, lane), xn);
    v8f hn = {};
    hn = wmma_f16(ah0, load_bT_frag(whh, 2 * RH + c, RH, 0, lane), hn);
    hn = wmma_f16(ah1, load_bT_frag(whh, 2 * RH + c, RH, 32, lane), hn);

    int col = c + col16;
    float br  = bih[col] + bhh[col];
    float bz  = bih[RH + col] + bhh[RH + col];
    float bxn = bih[2 * RH + col];
    float bhn = bhh[2 * RH + col];
#pragma unroll
    for (int v = 0; v < 8; ++v) {
      int row = m0 + 8 * half + v;
      float hold = h32[(size_t)row * RH + col];
      float r = 1.0f / (1.0f + __expf(-(sr[v] + br)));
      float z = 1.0f / (1.0f + __expf(-(sz[v] + bz)));
      float cand = tanhf(xn[v] + bxn + r * (hn[v] + bhn));
      float hnew = (1.0f - z) * cand + z * hold;
      h32[(size_t)row * RH + col] = hnew;
      h16[(size_t)row * RH + col] = (_Float16)hnew;
    }
  }
}

// ---------------------------------------------------------------------------
// MLP head (WMMA): hid = relu(h @ mlp_w1^T + b1); out = hid @ w2 + b2.
// One wave per 16-row tile: 4 col-tiles x 2 WMMAs (K=64), then a 16-lane
// butterfly reduction for the final 64-wide dot product.
// ---------------------------------------------------------------------------
__global__ void k_mlp(const _Float16* __restrict__ h16,
                      const _Float16* __restrict__ mw1,  // [64,64]
                      const float* __restrict__ b1,
                      const float* __restrict__ w2,      // [64]
                      const float* __restrict__ b2,
                      float* __restrict__ out, int N) {
  int lane = threadIdx.x & 31;
  int tile = blockIdx.x * (blockDim.x >> 5) + (threadIdx.x >> 5);
  int m0 = tile * 16;
  if (m0 >= N) return;
  int col16 = lane & 15, half = lane >> 4;

  v16h a0 = load_a_frag(h16, m0, RH, 0, lane);
  v16h a1 = load_a_frag(h16, m0, RH, 32, lane);

  float p[8];
#pragma unroll
  for (int v = 0; v < 8; ++v) p[v] = 0.0f;

#pragma unroll
  for (int j = 0; j < 4; ++j) {
    v8f c = {};
    c = wmma_f16(a0, load_bT_frag(mw1, j * 16, RH, 0, lane), c);
    c = wmma_f16(a1, load_bT_frag(mw1, j * 16, RH, 32, lane), c);
    int col = j * 16 + col16;
    float bb = b1[col], ww = w2[col];
#pragma unroll
    for (int v = 0; v < 8; ++v) p[v] += fmaxf(c[v] + bb, 0.0f) * ww;
  }
  // reduce over the 16 lanes holding the same rows (xor masks < 16 stay in group)
#pragma unroll
  for (int off = 1; off < 16; off <<= 1) {
#pragma unroll
    for (int v = 0; v < 8; ++v) p[v] += __shfl_xor(p[v], off, 32);
  }
  if (col16 == 0) {
#pragma unroll
    for (int v = 0; v < 8; ++v) out[m0 + 8 * half + v] = p[v] + b2[0];
  }
}

// ---------------------------------------------------------------------------
// Host launcher
// ---------------------------------------------------------------------------
extern "C" void kernel_launch(void* const* d_in, const int* in_sizes, int n_in,
                              void* d_out, int out_size, void* d_ws, size_t ws_size,
                              hipStream_t stream) {
  const float* dyn   = (const float*)d_in[0];
  const float* stat  = (const float*)d_in[1];
  const int*   ei    = (const int*)d_in[2];
  const float* w0    = (const float*)d_in[3];
  const float* b0    = (const float*)d_in[4];
  const float* w1    = (const float*)d_in[5];
  const float* b1g   = (const float*)d_in[6];
  const float* wih   = (const float*)d_in[7];
  const float* whh   = (const float*)d_in[8];
  const float* bih   = (const float*)d_in[9];
  const float* bhh   = (const float*)d_in[10];
  const float* mw1   = (const float*)d_in[11];
  const float* mb1   = (const float*)d_in[12];
  const float* mw2   = (const float*)d_in[13];
  const float* mb2   = (const float*)d_in[14];
  float* out = (float*)d_out;

  const int N = in_sizes[1] / STAT;
  const int T = in_sizes[0] / N;
  const int E = in_sizes[2] / 2;

  // workspace layout (256B aligned slices)
  char* base = (char*)d_ws;
  size_t off = 0;
  auto take = [&](size_t bytes) -> char* {
    char* p = base + off;
    off = (off + bytes + 255) & ~(size_t)255;
    return p;
  };
  float*     dinv   = (float*)take((size_t)N * 4);
  float*     pre    = (float*)take((size_t)N * GH * 4);
  float*     agg    = (float*)take((size_t)N * GH * 4);
  _Float16*  h0f16  = (_Float16*)take((size_t)N * GH * 2);
  _Float16*  htf16  = (_Float16*)take((size_t)N * GH * 2);
  float*     h32    = (float*)take((size_t)N * RH * 4);
  _Float16*  h16    = (_Float16*)take((size_t)N * RH * 2);
  _Float16*  w1Tf16 = (_Float16*)take(GH * GH * 2);
  _Float16*  wih16  = (_Float16*)take(3 * RH * GH * 2);
  _Float16*  whh16  = (_Float16*)take(3 * RH * RH * 2);
  _Float16*  mw1f16 = (_Float16*)take(RH * RH * 2);
  (void)ws_size; (void)n_in; (void)out_size;

  const int B = 256;
  auto g = [](int n, int b) { return (n + b - 1) / b; };
  const int ntiles = (N + 15) / 16;
  const int wmma_grid = (ntiles + 3) / 4;  // 4 waves / 128-thread block

  // --- one-time prep -------------------------------------------------------
  k_fill_f32<<<g(N, B), B, 0, stream>>>(dinv, 1.0f, N);            // self-loop
  k_degree<<<g(E, B), B, 0, stream>>>(ei, dinv, E);
  k_dinv<<<g(N, B), B, 0, stream>>>(dinv, N);
  k_fill_f32<<<g(N * RH, B), B, 0, stream>>>(h32, 0.0f, N * RH);   // h0 = 0
  k_fill_f16_zero<<<g(N * RH, B), B, 0, stream>>>(h16, N * RH);
  k_cvt_f16_T<<<g(GH * GH, B), B, 0, stream>>>(w1, w1Tf16, GH, GH);
  k_cvt_f16<<<g(3 * RH * GH, B), B, 0, stream>>>(wih, wih16, 3 * RH * GH);
  k_cvt_f16<<<g(3 * RH * RH, B), B, 0, stream>>>(whh, whh16, 3 * RH * RH);
  k_cvt_f16<<<g(RH * RH, B), B, 0, stream>>>(mw1, mw1f16, RH * RH);

  // --- per-timestep pipeline ----------------------------------------------
  for (int t = 0; t < T; ++t) {
    // GCN layer 0
    k_fill_f32<<<g(N * GH, B), B, 0, stream>>>(agg, 0.0f, N * GH);
    k_gcn_transform0<<<g(N, B), B, 0, stream>>>(dyn, stat, w0, pre, N, t);
    k_gcn_aggregate<<<g(E * 4, B), B, 0, stream>>>(ei, dinv, pre, agg, E);
    k_gcn_finish<<<g(N * GH, B), B, 0, stream>>>(agg, pre, dinv, b0, h0f16, N);
    // GCN layer 1 (WMMA transform)
    k_fill_f32<<<g(N * GH, B), B, 0, stream>>>(agg, 0.0f, N * GH);
    k_gcn_transform1<<<wmma_grid, 128, 0, stream>>>(h0f16, w1Tf16, pre, N);
    k_gcn_aggregate<<<g(E * 4, B), B, 0, stream>>>(ei, dinv, pre, agg, E);
    k_gcn_finish<<<g(N * GH, B), B, 0, stream>>>(agg, pre, dinv, b1g, htf16, N);
    // GRU step (WMMA)
    k_gru_step<<<wmma_grid, 128, 0, stream>>>(htf16, wih16, whh16, bih, bhh,
                                              h32, h16, N);
  }

  // --- MLP head (WMMA) -----------------------------------------------------
  k_mlp<<<wmma_grid, 128, 0, stream>>>(h16, mw1f16, mb1, mw2, mb2, out, N);
}
